// MiniAttentionQHead_17188459118911
// MI455X (gfx1250) — matrix operand order repaired and checked by reading, not compiled, not run on hardware
//
#include <hip/hip_runtime.h>

// ---------- vector types ----------
typedef __bf16        v16bf __attribute__((ext_vector_type(16)));
typedef float         v8f   __attribute__((ext_vector_type(8)));
typedef unsigned int  u32x4 __attribute__((ext_vector_type(4)));
typedef unsigned int  u32x2 __attribute__((ext_vector_type(2)));
typedef int           i32x4 __attribute__((ext_vector_type(4)));
typedef float         f32x4 __attribute__((ext_vector_type(4)));

union FragBF { v16bf v; u32x4 q[2]; };

#ifndef __has_builtin
#define __has_builtin(x) 0
#endif
#if __has_builtin(__builtin_amdgcn_global_load_async_to_lds_b128)
#define HAVE_ASYNC 1
#else
#define HAVE_ASYNC 0
#endif

#define GAS __attribute__((address_space(1)))
#define LAS __attribute__((address_space(3)))

template <int N>
__device__ __forceinline__ void wait_async() {
#if __has_builtin(__builtin_amdgcn_s_wait_asynccnt)
    __builtin_amdgcn_s_wait_asynccnt(N);
#else
    asm volatile("s_wait_asynccnt %0" :: "i"(N) : "memory");
#endif
}

__device__ __forceinline__ unsigned short f2bf(float x) {
    unsigned int u = __float_as_uint(x);
    u += 0x7FFFu + ((u >> 16) & 1u);          // round-to-nearest-even
    return (unsigned short)(u >> 16);
}
__device__ __forceinline__ float bf2f(unsigned short h) {
    return __uint_as_float(((unsigned int)h) << 16);
}
__device__ __forceinline__ void load_bf4(const unsigned short* p, float* o) {
    u32x2 r = *(const u32x2*)p;
    o[0] = bf2f((unsigned short)(r[0] & 0xFFFFu));
    o[1] = bf2f((unsigned short)(r[0] >> 16));
    o[2] = bf2f((unsigned short)(r[1] & 0xFFFFu));
    o[3] = bf2f((unsigned short)(r[1] >> 16));
}

// ---------- problem constants ----------
// B=4096, H=2048, NH=16, D=128, W=8, A=2, 3H=6144, K=2048

// ---------- conversion kernels ----------
__global__ __launch_bounds__(256)
void k_conv_hidden(const float* __restrict__ x, unsigned short* __restrict__ y, int n4) {
    int i = blockIdx.x * 256 + threadIdx.x;
    if (i >= n4) return;
    f32x4 v = *(const f32x4*)(x + (long)i * 4);
    u32x2 p;
    p[0] = (unsigned)f2bf(v[0]) | ((unsigned)f2bf(v[1]) << 16);
    p[1] = (unsigned)f2bf(v[2]) | ((unsigned)f2bf(v[3]) << 16);
    *(u32x2*)(y + (long)i * 4) = p;
}

__global__ __launch_bounds__(256)
void k_conv_buf(const float* __restrict__ ctx, const float* __restrict__ hid,
                const int* __restrict__ cptr, unsigned short* __restrict__ y) {
    int i  = blockIdx.x * 256 + threadIdx.x;   // over B*W*H/4 = 16,777,216
    int k4 = i & 511;                          // H/4 = 512
    int w  = (i >> 9) & 7;                     // W = 8
    int b  = i >> 12;
    int ptr = (((*cptr) % 8) + 8) % 8;
    const float* src = (w == ptr) ? (hid + (long)b * 2048 + (long)k4 * 4)
                                  : (ctx + ((long)b * 8 + w) * 2048 + (long)k4 * 4);
    f32x4 v = *(const f32x4*)src;
    u32x2 p;
    p[0] = (unsigned)f2bf(v[0]) | ((unsigned)f2bf(v[1]) << 16);
    p[1] = (unsigned)f2bf(v[2]) | ((unsigned)f2bf(v[3]) << 16);
    *(u32x2*)(y + (long)i * 4) = p;
}

// w_qkv [6144][2048] fp32 -> K-major bf16 [2048][6144], LDS-tiled transpose
__global__ __launch_bounds__(256)
void k_conv_w_T(const float* __restrict__ w, unsigned short* __restrict__ wt) {
    __shared__ float tile[32][33];
    int n0 = blockIdx.x * 32, k0 = blockIdx.y * 32;
    int tx = threadIdx.x & 31, ty = threadIdx.x >> 5;   // ty 0..7
#pragma unroll
    for (int r = 0; r < 4; ++r)
        tile[ty + r * 8][tx] = w[(long)(n0 + ty + r * 8) * 2048 + k0 + tx];
    __syncthreads();
#pragma unroll
    for (int r = 0; r < 4; ++r)
        wt[(long)(k0 + ty + r * 8) * 6144 + n0 + tx] = f2bf(tile[tx][ty + r * 8]);
}

// ---------- tiled bf16 WMMA GEMM:  C[M,N] = A[M,K] * B[K,N]  (B is K-major) ----------
#define BM 128
#define BN 128
#define BK 32
#define LDA_S 40     // shorts per sA row  (80 B, 16B aligned, bank stride 20 dw)
#define LDB_S 136    // shorts per sB row  (272 B, 16B aligned, bank stride 68 dw)
#define SA_BUF (BM * LDA_S)   // 5120 shorts
#define SB_BUF (BK * LDB_S)   // 4352 shorts

__global__ __launch_bounds__(256)
void k_gemm_bf16(const unsigned short* __restrict__ A, int lda,
                 const unsigned short* __restrict__ B, int ldb,
                 unsigned short* __restrict__ C, int ldc, int K) {
    __shared__ unsigned short sA[3 * SA_BUF];   // 30.0 KB
    __shared__ unsigned short sB[3 * SB_BUF];   // 25.5 KB

    const int  tid   = threadIdx.x;
    const int  lane  = tid & 31;
    const int  wave  = tid >> 5;
    const int  m0w   = (wave >> 2) * 64;   // wave M origin in block tile
    const int  n0w   = (wave & 3) * 32;    // wave N origin in block tile
    const long blockM = (long)blockIdx.y * BM;
    const long blockN = (long)blockIdx.x * BN;

    // global->LDS staging assignment (two 16B chunks per thread per tile)
    const int ca_row = tid >> 2;           // 0..63, +64 for second chunk
    const int ca_seg = (tid & 3) * 8;      // shorts
    const int cb_row = tid >> 4;           // 0..15, +16 for second chunk
    const int cb_seg = (tid & 15) * 8;     // shorts

    // running global pointers (advance by BK / BK*ldb each stage)
    const unsigned short* gA0 = A + (blockM + ca_row) * lda + ca_seg;
    const unsigned short* gA1 = A + (blockM + ca_row + 64) * lda + ca_seg;
    const unsigned short* gB0 = B + (long)cb_row * ldb + blockN + cb_seg;
    const unsigned short* gB1 = B + (long)(cb_row + 16) * ldb + blockN + cb_seg;
    const long bstep = (long)BK * ldb;

    // LDS destination offsets (fixed per thread, per buffer)
    const int la0 = ca_row * LDA_S + ca_seg;
    const int la1 = (ca_row + 64) * LDA_S + ca_seg;
    const int lb0 = cb_row * LDB_S + cb_seg;
    const int lb1 = (cb_row + 16) * LDB_S + cb_seg;

    auto stage = [&](int buf) {
#if HAVE_ASYNC
        __builtin_amdgcn_global_load_async_to_lds_b128(
            (GAS i32x4*)gA0, (LAS i32x4*)&sA[buf * SA_BUF + la0], 0, 0);
        __builtin_amdgcn_global_load_async_to_lds_b128(
            (GAS i32x4*)gA1, (LAS i32x4*)&sA[buf * SA_BUF + la1], 0, 0);
        __builtin_amdgcn_global_load_async_to_lds_b128(
            (GAS i32x4*)gB0, (LAS i32x4*)&sB[buf * SB_BUF + lb0], 0, 0);
        __builtin_amdgcn_global_load_async_to_lds_b128(
            (GAS i32x4*)gB1, (LAS i32x4*)&sB[buf * SB_BUF + lb1], 0, 0);
#else
        u32x4 a0 = *(const u32x4*)gA0;
        u32x4 a1 = *(const u32x4*)gA1;
        u32x4 b0 = *(const u32x4*)gB0;
        u32x4 b1 = *(const u32x4*)gB1;
        *(u32x4*)&sA[buf * SA_BUF + la0] = a0;
        *(u32x4*)&sA[buf * SA_BUF + la1] = a1;
        *(u32x4*)&sB[buf * SB_BUF + lb0] = b0;
        *(u32x4*)&sB[buf * SB_BUF + lb1] = b1;
#endif
        gA0 += BK; gA1 += BK; gB0 += bstep; gB1 += bstep;
    };

    v8f acc[4][2];
#pragma unroll
    for (int mt = 0; mt < 4; ++mt)
#pragma unroll
        for (int nt = 0; nt < 2; ++nt)
            acc[mt][nt] = (v8f){0.f, 0.f, 0.f, 0.f, 0.f, 0.f, 0.f, 0.f};

    // per-lane fragment addressing (ISA 7.12.2 layouts)
    const int mrow = lane & 15;            // A: lane -> M row
    const int kb2  = (lane >> 4) << 3;     // A: K sub-base 0 or 8
    const int ncol = lane & 15;            // B: lane -> N col, K sub-base kb2

    auto compute = [&](int cb) {
        FragBF afrag[4], bfrag[2];
#pragma unroll
        for (int mt = 0; mt < 4; ++mt) {
            int base = cb * SA_BUF + (m0w + mt * 16 + mrow) * LDA_S + kb2;
            afrag[mt].q[0] = *(const u32x4*)&sA[base];
            afrag[mt].q[1] = *(const u32x4*)&sA[base + 16];
        }
#pragma unroll
        for (int nt = 0; nt < 2; ++nt) {
            int base = cb * SB_BUF + n0w + nt * 16 + ncol;
#pragma unroll
            for (int i = 0; i < 8; ++i) {
                int k = (i < 4) ? (kb2 + 2 * i) : (16 + kb2 + 2 * (i - 4));
                unsigned lo = sB[base + k * LDB_S];
                unsigned hi = sB[base + (k + 1) * LDB_S];
                bfrag[nt].q[i >> 2][i & 3] = lo | (hi << 16);
            }
        }
#pragma unroll
        for (int mt = 0; mt < 4; ++mt)
#pragma unroll
            for (int nt = 0; nt < 2; ++nt)
                acc[mt][nt] = __builtin_amdgcn_wmma_f32_16x16x32_bf16(
                    false, afrag[mt].v, false, bfrag[nt].v,
                    (short)0, acc[mt][nt], false, false);
    };

    const int nk = K / BK;
    // pipeline prologue: two tiles in flight
    stage(0);
    if (nk > 1) stage(1);

    int cur = 0;      // buffer consumed this iteration  (kt   % 3)
    int nxt = 2;      // buffer staged this iteration    (kt+2 % 3)

    // steady state: 2 tiles always in flight; drain only the oldest 4 ops
    for (int kt = 0; kt + 2 < nk; ++kt) {
#if HAVE_ASYNC
        wait_async<4>();
#endif
        __syncthreads();   // tile kt visible to all; all waves done reading tile kt-1
        stage(nxt);
        compute(cur);
        cur = (cur == 2) ? 0 : cur + 1;
        nxt = (nxt == 2) ? 0 : nxt + 1;
    }
    // tail kt = nk-2: one younger tile still in flight
    if (nk >= 2) {
#if HAVE_ASYNC
        wait_async<4>();
#endif
        __syncthreads();
        compute(cur);
        cur = (cur == 2) ? 0 : cur + 1;
    }
    // tail kt = nk-1: drain everything
#if HAVE_ASYNC
    wait_async<0>();
#endif
    __syncthreads();
    compute(cur);

    // epilogue: C/D layout — lane&15 = N col, VGPR r -> M = r + (lane>=16 ? 8 : 0)
    const int  col = lane & 15;
    const int  mhi = (lane >> 4) * 8;
#pragma unroll
    for (int mt = 0; mt < 4; ++mt)
#pragma unroll
        for (int nt = 0; nt < 2; ++nt) {
            long crow0 = blockM + m0w + mt * 16 + mhi;
            long cc    = blockN + n0w + nt * 16 + col;
#pragma unroll
            for (int r = 0; r < 8; ++r)
                C[(crow0 + r) * ldc + cc] = f2bf(acc[mt][nt][r]);
        }
}

// ---------- attention: one wave per (batch, head), 9 keys ----------
__global__ __launch_bounds__(256)
void k_attn(const unsigned short* __restrict__ qkv,   // [B][6144] bf16: q|k|v
            const unsigned short* __restrict__ kv,    // [B*W][4096] bf16: k|v
            float* __restrict__ att) {                // [B][2048] fp32
    int gw   = blockIdx.x * 8 + (threadIdx.x >> 5);
    int lane = threadIdx.x & 31;
    int b = gw >> 4, h = gw & 15;
    const long qoff = (long)b * 6144 + h * 128 + lane * 4;

    float q[4];
    load_bf4(qkv + qoff, q);

    float s[9];
#pragma unroll
    for (int j = 0; j < 9; ++j) {
        const unsigned short* kp = (j == 0)
            ? qkv + (long)b * 6144 + 2048 + h * 128 + lane * 4
            : kv + ((long)b * 8 + (j - 1)) * 4096 + h * 128 + lane * 4;
        float kj[4];
        load_bf4(kp, kj);
        float d = q[0] * kj[0] + q[1] * kj[1] + q[2] * kj[2] + q[3] * kj[3];
        d += __shfl_xor(d, 1, 32);
        d += __shfl_xor(d, 2, 32);
        d += __shfl_xor(d, 4, 32);
        d += __shfl_xor(d, 8, 32);
        d += __shfl_xor(d, 16, 32);
        s[j] = d * 0.08838834764831845f;   // 1/sqrt(128)
    }
    float mx = s[0];
#pragma unroll
    for (int j = 1; j < 9; ++j) mx = fmaxf(mx, s[j]);
    float p[9], den = 0.f;
#pragma unroll
    for (int j = 0; j < 9; ++j) { p[j] = __expf(s[j] - mx); den += p[j]; }
    float inv = 1.0f / den;

    float o0 = 0.f, o1 = 0.f, o2 = 0.f, o3 = 0.f;
#pragma unroll
    for (int j = 0; j < 9; ++j) {
        const unsigned short* vp = (j == 0)
            ? qkv + (long)b * 6144 + 4096 + h * 128 + lane * 4
            : kv + ((long)b * 8 + (j - 1)) * 4096 + 2048 + h * 128 + lane * 4;
        float vj[4];
        load_bf4(vp, vj);
        float w = p[j] * inv;
        o0 += w * vj[0]; o1 += w * vj[1]; o2 += w * vj[2]; o3 += w * vj[3];
    }
    f32x4 out; out[0] = o0; out[1] = o1; out[2] = o2; out[3] = o3;
    *(f32x4*)(att + (long)b * 2048 + h * 128 + lane * 4) = out;
}

// ---------- output head: out[b][a] = att[b]·w_out[a] + b_out[a], A=2 ----------
__global__ __launch_bounds__(256)
void k_final(const float* __restrict__ att, const float* __restrict__ wout,
             const float* __restrict__ bout, float* __restrict__ out) {
    int b    = blockIdx.x * 8 + (threadIdx.x >> 5);
    int lane = threadIdx.x & 31;
    float s0 = 0.f, s1 = 0.f;
    for (int i = lane * 4; i < 2048; i += 128) {
        f32x4 x  = *(const f32x4*)(att + (long)b * 2048 + i);
        f32x4 w0 = *(const f32x4*)(wout + i);
        f32x4 w1 = *(const f32x4*)(wout + 2048 + i);
        s0 += x[0] * w0[0] + x[1] * w0[1] + x[2] * w0[2] + x[3] * w0[3];
        s1 += x[0] * w1[0] + x[1] * w1[1] + x[2] * w1[2] + x[3] * w1[3];
    }
#pragma unroll
    for (int m = 1; m < 32; m <<= 1) {
        s0 += __shfl_xor(s0, m, 32);
        s1 += __shfl_xor(s1, m, 32);
    }
    if (lane == 0) {
        out[(long)b * 2 + 0] = s0 + bout[0];
        out[(long)b * 2 + 1] = s1 + bout[1];
    }
}

// ---------- launch ----------
extern "C" void kernel_launch(void* const* d_in, const int* in_sizes, int n_in,
                              void* d_out, int out_size, void* d_ws, size_t ws_size,
                              hipStream_t stream) {
    (void)in_sizes; (void)n_in; (void)out_size; (void)ws_size;
    const float* hs   = (const float*)d_in[0];
    const float* ctx  = (const float*)d_in[1];
    const float* wqkv = (const float*)d_in[2];
    const float* wout = (const float*)d_in[3];
    const float* bout = (const float*)d_in[4];
    const int*   cptr = (const int*)d_in[5];

    char* ws = (char*)d_ws;
    unsigned short* hid_bf = (unsigned short*)(ws);               //  16.8 MB
    unsigned short* buf_bf = (unsigned short*)(ws + 16777216L);   // 134.2 MB
    unsigned short* w_bf   = (unsigned short*)(ws + 150994944L);  //  25.2 MB (K-major 2048x6144)
    unsigned short* qkv_bf = (unsigned short*)(ws + 176160768L);  //  50.3 MB (4096x6144)
    unsigned short* kv_bf  = (unsigned short*)(ws + 226492416L);  // 268.4 MB (32768x4096)
    float*          att    = (float*)(ws + 494927872L);           //  33.6 MB (4096x2048)
    float* out = (float*)d_out;

    k_conv_hidden<<<8192, 256, 0, stream>>>(hs, hid_bf, 2097152);
    k_conv_buf<<<65536, 256, 0, stream>>>(ctx, hs, cptr, buf_bf);
    k_conv_w_T<<<dim3(192, 64), 256, 0, stream>>>(wqkv, w_bf);

    // GEMM1: qkv = hidden @ w_qkv^T   (M=4096, N=6144, K=2048)
    k_gemm_bf16<<<dim3(48, 32), 256, 0, stream>>>(hid_bf, 2048, w_bf, 6144,
                                                  qkv_bf, 6144, 2048);
    // GEMM2: ctx k,v = buf @ w_kv^T   (M=32768, N=4096, K=2048)
    k_gemm_bf16<<<dim3(32, 256), 256, 0, stream>>>(buf_bf, 2048, w_bf + 2048, 6144,
                                                   kv_bf, 4096, 2048);

    k_attn<<<8192, 256, 0, stream>>>(qkv_bf, kv_bf, att);
    k_final<<<512, 256, 0, stream>>>(att, wout, bout, out);
}
